// RSWABlock_67095979099131
// MI455X (gfx1250) — compile-verified, matrix operand order checked by reading, not run on hardware
//
#include <hip/hip_runtime.h>
#include <hip/hip_bf16.h>
#include <math.h>

typedef __attribute__((ext_vector_type(16))) __bf16 v16bf;
typedef __attribute__((ext_vector_type(8)))  float  v8f;

#define DIMC   192
#define HWSZ   16384      // 128*128
#define NWIN   4096       // 16 * 16 * 16 windows
#define ATT_SCALE 0.17677669529663687f   // 32^-0.5

__device__ __forceinline__ unsigned short f2bf(float f) {
  unsigned u = __float_as_uint(f);
  u += 0x7fffu + ((u >> 16) & 1u);       // round-to-nearest-even
  return (unsigned short)(u >> 16);
}
__device__ __forceinline__ float bf2f(unsigned short h) {
  return __uint_as_float(((unsigned)h) << 16);
}
__device__ __forceinline__ v8f vzero8() {
  v8f z;
#pragma unroll
  for (int g = 0; g < 8; ++g) z[g] = 0.0f;
  return z;
}

// CDNA5 async copy: 16 bytes/lane, global -> LDS, tracked by ASYNCcnt.
// LDS address = low 32 bits of the generic pointer (LDS aperture offset).
__device__ __forceinline__ void async_g2l_b128(void* lds_ptr, const void* gptr) {
  unsigned lofs = (unsigned)(uintptr_t)lds_ptr;
  unsigned long long ga = (unsigned long long)(uintptr_t)gptr;
  asm volatile("global_load_async_to_lds_b128 %0, %1, off"
               :: "v"(lofs), "v"(ga) : "memory");
}
__device__ __forceinline__ void wait_async0() {
  asm volatile("s_wait_asynccnt 0x0" ::: "memory");
}

// A/B fragment load (16x32 bf16, MxK) from row-major bf16 with leading dim `ld`
// (elements).  Per CDNA5 ISA: lanes 0-15 hold rows M=0..15 with K = {kb..kb+7,
// kb+16..kb+23}; lanes 16-31 hold the same rows with K shifted by +8.
// B operands are stored transposed (N x K row-major) so the same loader works.
__device__ __forceinline__ v16bf load_frag(const unsigned short* base, int ld,
                                           int row0, int k0) {
  int lane = threadIdx.x & 31;
  int r = row0 + (lane & 15);
  int kb = k0 + ((lane >> 4) << 3);
  union { uint4 u[2]; v16bf v; } t;
  t.u[0] = *(const uint4*)(base + r * ld + kb);
  t.u[1] = *(const uint4*)(base + r * ld + kb + 16);
  return t.v;
}

__device__ __forceinline__ v8f wmma_bf16(v16bf a, v16bf b, v8f c) {
  return __builtin_amdgcn_wmma_f32_16x16x32_bf16(false, a, false, b, (short)0, c,
                                                 false, false);
}

// ---------------------------------------------------------------------------
// Kernel 0: weight prep — bf16 convert, transpose B operands to (N,K) layout.
// ---------------------------------------------------------------------------
__global__ __launch_bounds__(256) void prep_kernel(
    const float* __restrict__ wpre, const float* __restrict__ wqkv,
    const float* __restrict__ w1,   const float* __restrict__ w2,
    const float* __restrict__ wproj,
    unsigned short* __restrict__ dwpre,  unsigned short* __restrict__ dwqkvT,
    unsigned short* __restrict__ dw1T,   unsigned short* __restrict__ dw2T,
    unsigned short* __restrict__ dwproj) {
  int i = blockIdx.x * 256 + threadIdx.x;
  if (i < 192 * 192) dwpre[i] = f2bf(wpre[i]);                 // (o,c) == B^T
  if (i < 576 * 192) {                                          // w_qkv (192,576) -> (576,192)
    int r = i / 192, k = i % 192;
    dwqkvT[i] = f2bf(wqkv[k * 576 + r]);
  }
  if (i < 768 * 192) {                                          // w1 (192,768) -> (768,192)
    int r = i / 192, k = i % 192;
    dw1T[i] = f2bf(w1[k * 768 + r]);
  }
  if (i < 192 * 768) {                                          // w2 (768,192) -> (192,768)
    int r = i / 768, k = i % 768;
    dw2T[i] = f2bf(w2[k * 192 + r]);
  }
  if (i < 192 * 192) dwproj[i] = f2bf(wproj[i]);               // (o,c) == B^T
}

// ---------------------------------------------------------------------------
// Kernel 1: 1x1 conv  h1[m, o] = sum_c x[b,c,hw] * w_pre1[o,c] + b_pre1[o]
// One block = 64 raster tokens, 8 waves, tile grid 4(M) x 12(N), 6 K-steps.
// ---------------------------------------------------------------------------
__global__ __launch_bounds__(256) void pre1_kernel(
    const float* __restrict__ x, const unsigned short* __restrict__ wPre,
    const float* __restrict__ bPre, float* __restrict__ h1) {
  __shared__ __align__(16) unsigned short sA[64 * 40];
  const int tid  = threadIdx.x;
  const int lane = tid & 31;
  const int wave = tid >> 5;
  const int cn   = lane & 15;
  const int rofs = (lane >> 4) << 3;
  const int mt   = wave >> 1;
  const int nt0  = (wave & 1) * 6;
  const int m0   = blockIdx.x * 64;
  const int b    = m0 >> 14;
  const int hw0  = m0 & (HWSZ - 1);
  const float* xb = x + (size_t)b * DIMC * HWSZ + hw0;

  v8f acc[6];
#pragma unroll
  for (int i = 0; i < 6; ++i) acc[i] = vzero8();

  for (int kk = 0; kk < 6; ++kk) {
    __syncthreads();
#pragma unroll
    for (int e = 0; e < 8; ++e) {            // stage 64 tokens x 32 channels
      int idx = tid + e * 256;
      int j = idx >> 6, t = idx & 63;
      sA[t * 40 + j] = f2bf(xb[(size_t)(kk * 32 + j) * HWSZ + t]);
    }
    __syncthreads();
    v16bf a = load_frag(sA, 40, mt * 16, 0);
    v16bf bfr[6];
#pragma unroll
    for (int i = 0; i < 6; ++i)
      bfr[i] = load_frag(wPre + (size_t)((nt0 + i) * 16) * DIMC, DIMC, 0, kk * 32);
#pragma unroll
    for (int i = 0; i < 6; ++i) acc[i] = wmma_bf16(a, bfr[i], acc[i]);
  }
#pragma unroll
  for (int i = 0; i < 6; ++i) {
    int col = (nt0 + i) * 16 + cn;
    float bias = bPre[col];
#pragma unroll
    for (int g = 0; g < 8; ++g) {
      int row = mt * 16 + g + rofs;
      h1[(size_t)(m0 + row) * DIMC + col] = acc[i][g] + bias;
    }
  }
}

// ---------------------------------------------------------------------------
// Kernel 2: depthwise 3x3 (SAME, zero pad) + channel LayerNorm, output p in
// bf16, *window-token* order: token = n*64 + t.
// One 192-thread block per pixel; thread = channel.
// ---------------------------------------------------------------------------
__global__ __launch_bounds__(192) void dwln_kernel(
    const float* __restrict__ h1, const float* __restrict__ wdw,
    const float* __restrict__ bdw, const float* __restrict__ lng,
    const float* __restrict__ lnb, unsigned short* __restrict__ gP) {
  const int m = blockIdx.x;
  const int c = threadIdx.x;
  const int b = m >> 14, hw = m & (HWSZ - 1);
  const int y = hw >> 7, xx = hw & 127;

  float acc = bdw[c];
  const float* w9 = wdw + c * 9;
#pragma unroll
  for (int dy = 0; dy < 3; ++dy) {
    int yy = y + dy - 1;
    if (yy < 0 || yy > 127) continue;
#pragma unroll
    for (int dx = 0; dx < 3; ++dx) {
      int x2 = xx + dx - 1;
      if (x2 < 0 || x2 > 127) continue;
      acc += w9[dy * 3 + dx] *
             h1[(size_t)((b << 14) + yy * 128 + x2) * DIMC + c];
    }
  }
  // LayerNorm over 192 channels
  __shared__ float red[12];
  float s = acc, s2 = acc * acc;
#pragma unroll
  for (int d = 16; d >= 1; d >>= 1) {
    s  += __shfl_xor(s, d);
    s2 += __shfl_xor(s2, d);
  }
  int wv = c >> 5;
  if ((c & 31) == 0) { red[wv] = s; red[6 + wv] = s2; }
  __syncthreads();
  float S = 0.f, S2 = 0.f;
#pragma unroll
  for (int i = 0; i < 6; ++i) { S += red[i]; S2 += red[6 + i]; }
  float mu  = S * (1.f / 192.f);
  float var = S2 * (1.f / 192.f) - mu * mu;
  float nv  = (acc - mu) * rsqrtf(var + 1e-6f) * lng[c] + lnb[c];

  int n = (b << 8) + ((y >> 3) << 4) + (xx >> 3);
  int t = ((y & 7) << 3) + (xx & 7);
  gP[(size_t)(n * 64 + t) * DIMC + c] = f2bf(nv);
}

// ---------------------------------------------------------------------------
// Kernel 3: fused per-window  QKV -> MHSA -> gMLP -> product.
// Block = 1 window (64 tokens), 256 threads = 8 wave32.
// ---------------------------------------------------------------------------
#define SM_P    0
#define SM_Q    25600
#define SM_K    51200
#define SM_VT   76800
#define SM_ATT  104448
#define SM_PW   130048
#define SM_TOT  146432

__global__ __launch_bounds__(256) void win_kernel(
    const unsigned short* __restrict__ gP,
    const unsigned short* __restrict__ wQkvT, const float* __restrict__ bQkv,
    const unsigned short* __restrict__ w1T,   const float* __restrict__ b1,
    const unsigned short* __restrict__ w2T,   const float* __restrict__ b2,
    unsigned short* __restrict__ gProd) {
  __shared__ __align__(16) unsigned char smem[SM_TOT];
  unsigned short* sP  = (unsigned short*)(smem + SM_P);    // 64x192, ld 200
  unsigned short* sQ  = (unsigned short*)(smem + SM_Q);    // 64x192, ld 200
  unsigned short* sK  = (unsigned short*)(smem + SM_K);    // 64x192, ld 200
  unsigned short* sVT = (unsigned short*)(smem + SM_VT);   // 192x64, ld 72 (V^T)
  unsigned short* sAt = (unsigned short*)(smem + SM_ATT);  // 64x192, ld 200 (attn out)
  unsigned short* sHid = sQ;                               // reuse for MLP hidden

  const int n    = blockIdx.x;
  const int tid  = threadIdx.x;
  const int lane = tid & 31;
  const int wave = tid >> 5;
  const int cn   = lane & 15;
  const int rofs = (lane >> 4) << 3;
  const int mt   = wave >> 1;
  const int nt0  = (wave & 1) * 6;
  unsigned short* sPW = (unsigned short*)(smem + SM_PW) + wave * 16 * 64;

  // --- phase 1: async-copy p tile (64x192 bf16) into LDS, 16B per lane ---
#pragma unroll
  for (int it = 0; it < 6; ++it) {          // 1536 chunks of 16B / 256 threads
    int q = tid + it * 256;
    int r = q / 24, c8 = (q % 24) * 8;
    async_g2l_b128(sP + r * 200 + c8,
                   gP + (size_t)(n * 64 + r) * DIMC + c8);
  }
  wait_async0();
  __syncthreads();

  // --- phase 2: QKV GEMM (64x192 @ 192x576) ---
  for (int comp = 0; comp < 3; ++comp) {
    v8f acc[6];
#pragma unroll
    for (int i = 0; i < 6; ++i) acc[i] = vzero8();
    for (int kk = 0; kk < 6; ++kk) {
      v16bf a = load_frag(sP, 200, mt * 16, kk * 32);
      v16bf bfr[6];
#pragma unroll
      for (int i = 0; i < 6; ++i)
        bfr[i] = load_frag(
            wQkvT + (size_t)(comp * 192 + (nt0 + i) * 16) * DIMC, DIMC, 0, kk * 32);
#pragma unroll
      for (int i = 0; i < 6; ++i) acc[i] = wmma_bf16(a, bfr[i], acc[i]);
    }
#pragma unroll
    for (int i = 0; i < 6; ++i) {
      int col = (nt0 + i) * 16 + cn;
      float bias = bQkv[comp * 192 + col];
#pragma unroll
      for (int g = 0; g < 8; ++g) {
        int row = mt * 16 + g + rofs;
        unsigned short hv = f2bf(acc[i][g] + bias);
        if      (comp == 0) sQ[row * 200 + col]  = hv;
        else if (comp == 1) sK[row * 200 + col]  = hv;
        else                sVT[col * 72 + row]  = hv;   // transposed V
      }
    }
  }
  __syncthreads();

  // --- phase 3: attention, 24 tasks = 6 heads x 4 row-blocks, 3 per wave ---
  for (int task = wave * 3; task < wave * 3 + 3; ++task) {
    int h  = task >> 2;
    int rb = task & 3;
    v16bf aq = load_frag(sQ, 200, rb * 16, h * 32);   // Q rows, K-depth = hd = 32
    v8f s[4];
#pragma unroll
    for (int nt = 0; nt < 4; ++nt) {
      v16bf bk = load_frag(sK, 200, nt * 16, h * 32); // B^T = K (token-major)
      s[nt] = wmma_bf16(aq, bk, vzero8());
    }
    // row softmax: cols live in {4 reg tiles} x {16 lanes of the half-wave}
#pragma unroll
    for (int g = 0; g < 8; ++g) {
      float vals[4];
      float mx = -1e30f;
#pragma unroll
      for (int nt = 0; nt < 4; ++nt) {
        vals[nt] = s[nt][g] * ATT_SCALE;
        mx = fmaxf(mx, vals[nt]);
      }
#pragma unroll
      for (int d = 8; d >= 1; d >>= 1) mx = fmaxf(mx, __shfl_xor(mx, d));
      float sum = 0.f;
#pragma unroll
      for (int nt = 0; nt < 4; ++nt) { vals[nt] = __expf(vals[nt] - mx); sum += vals[nt]; }
#pragma unroll
      for (int d = 8; d >= 1; d >>= 1) sum += __shfl_xor(sum, d);
      float inv = 1.f / sum;
#pragma unroll
      for (int nt = 0; nt < 4; ++nt)
        sPW[(g + rofs) * 64 + nt * 16 + cn] = f2bf(vals[nt] * inv);
    }
    // O = P (16x64) @ V (64x32)
#pragma unroll
    for (int nt2 = 0; nt2 < 2; ++nt2) {
      v8f o = vzero8();
#pragma unroll
      for (int kt = 0; kt < 2; ++kt) {
        v16bf ap = load_frag(sPW, 64, 0, kt * 32);
        v16bf bv = load_frag(sVT, 72, h * 32 + nt2 * 16, kt * 32);  // B^T = V^T
        o = wmma_bf16(ap, bv, o);
      }
#pragma unroll
      for (int g = 0; g < 8; ++g)
        sAt[(rb * 16 + g + rofs) * 200 + h * 32 + nt2 * 16 + cn] = f2bf(o[g]);
    }
  }
  __syncthreads();

  // --- phase 4: gMLP, hidden in 4 chunks of 192 (staged bf16 in sHid=sQ) ---
  v8f og[6];
#pragma unroll
  for (int i = 0; i < 6; ++i) og[i] = vzero8();
  for (int cc = 0; cc < 4; ++cc) {
    v8f ha[6];
#pragma unroll
    for (int i = 0; i < 6; ++i) ha[i] = vzero8();
    for (int kk = 0; kk < 6; ++kk) {
      v16bf a = load_frag(sP, 200, mt * 16, kk * 32);
      v16bf bfr[6];
#pragma unroll
      for (int i = 0; i < 6; ++i)
        bfr[i] = load_frag(
            w1T + (size_t)(cc * 192 + (nt0 + i) * 16) * DIMC, DIMC, 0, kk * 32);
#pragma unroll
      for (int i = 0; i < 6; ++i) ha[i] = wmma_bf16(a, bfr[i], ha[i]);
    }
#pragma unroll
    for (int i = 0; i < 6; ++i) {
      int col = (nt0 + i) * 16 + cn;
      float bias = b1[cc * 192 + col];
#pragma unroll
      for (int g = 0; g < 8; ++g) {
        float v = ha[i][g] + bias;
        float gl = 0.5f * v * (1.f + erff(v * 0.70710678118654752f)); // exact GELU
        sHid[(mt * 16 + g + rofs) * 200 + col] = f2bf(gl);
      }
    }
    __syncthreads();
    for (int kk2 = 0; kk2 < 6; ++kk2) {
      v16bf a = load_frag(sHid, 200, mt * 16, kk2 * 32);
      v16bf bfr[6];
#pragma unroll
      for (int i = 0; i < 6; ++i)
        bfr[i] = load_frag(w2T + (size_t)((nt0 + i) * 16) * 768, 768, 0,
                           cc * 192 + kk2 * 32);
#pragma unroll
      for (int i = 0; i < 6; ++i) og[i] = wmma_bf16(a, bfr[i], og[i]);
    }
    __syncthreads();
  }

  // --- phase 5: out = attn * gmlp, write bf16 window-token-major ---
#pragma unroll
  for (int i = 0; i < 6; ++i) {
    int col = (nt0 + i) * 16 + cn;
    float bias = b2[col];
#pragma unroll
    for (int g = 0; g < 8; ++g) {
      int row = mt * 16 + g + rofs;
      float v = (og[i][g] + bias) * bf2f(sAt[row * 200 + col]);
      gProd[(size_t)(n * 64 + row) * DIMC + col] = f2bf(v);
    }
  }
}

// ---------------------------------------------------------------------------
// Kernel 4: out projection + bias + residual, scatter back to NCHW fp32.
// ---------------------------------------------------------------------------
__global__ __launch_bounds__(256) void proj_kernel(
    const unsigned short* __restrict__ gProd,
    const unsigned short* __restrict__ wProj, const float* __restrict__ bProj,
    const float* __restrict__ x, float* __restrict__ out) {
  __shared__ __align__(16) unsigned short sA[64 * 200];
  const int n    = blockIdx.x;
  const int tid  = threadIdx.x;
  const int lane = tid & 31;
  const int wave = tid >> 5;
  const int cn   = lane & 15;
  const int rofs = (lane >> 4) << 3;
  const int mt   = wave >> 1;
  const int nt0  = (wave & 1) * 6;

  // async-copy prod tile into LDS, 16B per lane
#pragma unroll
  for (int it = 0; it < 6; ++it) {
    int q = tid + it * 256;
    int r = q / 24, c8 = (q % 24) * 8;
    async_g2l_b128(sA + r * 200 + c8,
                   gProd + (size_t)(n * 64 + r) * DIMC + c8);
  }
  wait_async0();
  __syncthreads();

  v8f acc[6];
#pragma unroll
  for (int i = 0; i < 6; ++i) acc[i] = vzero8();
  for (int kk = 0; kk < 6; ++kk) {
    v16bf a = load_frag(sA, 200, mt * 16, kk * 32);
    v16bf bfr[6];
#pragma unroll
    for (int i = 0; i < 6; ++i)
      bfr[i] = load_frag(wProj + (size_t)((nt0 + i) * 16) * DIMC, DIMC, 0, kk * 32);
#pragma unroll
    for (int i = 0; i < 6; ++i) acc[i] = wmma_bf16(a, bfr[i], acc[i]);
  }
  const int bI = n >> 8, rem = n & 255, wy = rem >> 4, wx = rem & 15;
#pragma unroll
  for (int i = 0; i < 6; ++i) {
    int o = (nt0 + i) * 16 + cn;
    float bias = bProj[o];
#pragma unroll
    for (int g = 0; g < 8; ++g) {
      int t = mt * 16 + g + rofs;
      int y = wy * 8 + (t >> 3), xx = wx * 8 + (t & 7);
      size_t adr = ((size_t)bI * DIMC + o) * HWSZ + y * 128 + xx;
      out[adr] = acc[i][g] + bias + x[adr];
    }
  }
}

// ---------------------------------------------------------------------------
extern "C" void kernel_launch(void* const* d_in, const int* in_sizes, int n_in,
                              void* d_out, int out_size, void* d_ws, size_t ws_size,
                              hipStream_t stream) {
  const float* x      = (const float*)d_in[0];
  const float* w_pre1 = (const float*)d_in[1];
  const float* b_pre1 = (const float*)d_in[2];
  const float* w_dw   = (const float*)d_in[3];
  const float* b_dw   = (const float*)d_in[4];
  const float* ln_g   = (const float*)d_in[5];
  const float* ln_b   = (const float*)d_in[6];
  const float* w_qkv  = (const float*)d_in[7];
  const float* b_qkv  = (const float*)d_in[8];
  const float* w1     = (const float*)d_in[9];
  const float* b1     = (const float*)d_in[10];
  const float* w2     = (const float*)d_in[11];
  const float* b2     = (const float*)d_in[12];
  const float* w_proj = (const float*)d_in[13];
  const float* b_proj = (const float*)d_in[14];

  char* ws = (char*)d_ws;
  unsigned short* wpre_b  = (unsigned short*)(ws + 0);          //  73728 B
  unsigned short* wqkvT   = (unsigned short*)(ws + 73728);      // 221184 B
  unsigned short* w1T     = (unsigned short*)(ws + 294912);     // 294912 B
  unsigned short* w2T     = (unsigned short*)(ws + 589824);     // 294912 B
  unsigned short* wproj_b = (unsigned short*)(ws + 884736);     //  73728 B
  float*          h1      = (float*)(ws + 958464);              // 201326592 B
  unsigned short* pbuf    = (unsigned short*)(ws + 202285056);  // 100663296 B
  unsigned short* prod    = (unsigned short*)(ws + 302948352);  // 100663296 B

  prep_kernel<<<576, 256, 0, stream>>>(w_pre1, w_qkv, w1, w2, w_proj,
                                       wpre_b, wqkvT, w1T, w2T, wproj_b);
  pre1_kernel<<<4096, 256, 0, stream>>>(x, wpre_b, b_pre1, h1);
  dwln_kernel<<<262144, 192, 0, stream>>>(h1, w_dw, b_dw, ln_g, ln_b, pbuf);
  win_kernel<<<NWIN, 256, 0, stream>>>(pbuf, wqkvT, b_qkv, w1T, b1, w2T, b2, prod);
  proj_kernel<<<NWIN, 256, 0, stream>>>(prod, wproj_b, b_proj, x, (float*)d_out);
}
